// FastAttention_31370441130035
// MI455X (gfx1250) — matrix operand order repaired and checked by reading, not compiled, not run on hardware
//
#include <hip/hip_runtime.h>

typedef _Float16 f16;
typedef __attribute__((ext_vector_type(16))) _Float16 v16h;
typedef __attribute__((ext_vector_type(8)))  float    v8f;

#define B_      4
#define H_      8
#define N_      4096
#define D_      64
#define NF      266          // real feature count
#define FP      288          // padded feature count (9 * 32)
#define CH      32           // chunk size == WMMA K for f16
#define NCHUNK  128
#define KEPS    1e-4f
#define AEPS    1e-6f
#define NORMZ   0.35355339059327373f   // 64^-0.25
#define RATIO   0.06131393394849658f   // 266^-0.5

#if defined(__has_builtin)
#if __has_builtin(__builtin_amdgcn_global_load_async_to_lds_b128) && \
    __has_builtin(__builtin_amdgcn_s_wait_asynccnt)
#define USE_ASYNC 1
#endif
#endif
#ifndef USE_ASYNC
#define USE_ASYNC 0
#endif

// ---------- monotonic float<->uint key for global max ----------
__device__ inline unsigned fkey(float x) {
  unsigned b = __float_as_uint(x);
  return (b & 0x80000000u) ? ~b : (b | 0x80000000u);
}
__device__ inline float funkey(unsigned kx) {
  unsigned b = (kx & 0x80000000u) ? (kx & 0x7fffffffu) : ~kx;
  return __uint_as_float(b);
}

// ---------- async global->LDS copy (16B per lane), ASYNCcnt-tracked ----------
#if USE_ASYNC
typedef int vi4 __attribute__((vector_size(4 * sizeof(int))));
typedef __attribute__((address_space(1))) vi4* gvi4_p;   // global 128b ptr
typedef __attribute__((address_space(3))) vi4* lvi4_p;   // LDS 128b ptr
#endif

__device__ inline void async_cp16(const float* g, float* l) {
#if USE_ASYNC
  __builtin_amdgcn_global_load_async_to_lds_b128((gvi4_p)(g), (lvi4_p)(l), 0, 0);
#else
  (void)g; (void)l;
#endif
}

// ---------- WMMA fragment helpers (CDNA5 wave32 layouts) ----------
// IS_A: A-matrix 16x32 packing: K = (i>>2)*16 + half*8 + (i&3)*2
// else: B-matrix 32x16 packing: K = (i>>2)*16 + half*8 + (i&3)*2
// KROW=false: element(lanedim L, K) at p[(L)*ld + K]
// KROW=true : element(lanedim L, K) at p[(K)*ld + L]
template <bool IS_A, bool KROW>
__device__ inline v16h ld_frag(const f16* __restrict__ p, int ld, int l0, int k0) {
  const int lane = threadIdx.x & 31;
  const int lr   = l0 + (lane & 15);
  const int half = lane >> 4;
  v16h r;
#pragma unroll
  for (int i = 0; i < 8; ++i) {
    int kk = IS_A ? (k0 + ((i >> 2) * 16) + half * 8 + ((i & 3) * 2))
                  : (k0 + ((i >> 2) * 16) + half * 8 + ((i & 3) * 2));
    if (KROW) {
      r[2 * i]     = p[(kk)     * ld + lr];
      r[2 * i + 1] = p[(kk + 1) * ld + lr];
    } else {
      r[2 * i]     = p[lr * ld + kk];
      r[2 * i + 1] = p[lr * ld + kk + 1];
    }
  }
  return r;
}

__device__ inline v8f wmma16(v16h a, v16h b, v8f c) {
  return __builtin_amdgcn_wmma_f32_16x16x32_f16(false, a, false, b, (short)0, c,
                                                false, false);
}

// C/D f32 tile: VGPR i holds row (r0 + i + 8*half), col = c0 + (lane&15)
__device__ inline void st_acc_f32(float* __restrict__ p, int ld, int r0, int c0, v8f a) {
  const int lane = threadIdx.x & 31;
  const int c = c0 + (lane & 15), half = lane >> 4;
#pragma unroll
  for (int i = 0; i < 8; ++i) p[(r0 + i + 8 * half) * ld + c] = a[i];
}
__device__ inline void st_acc_f16(f16* __restrict__ p, int ld, int r0, int c0, v8f a) {
  const int lane = threadIdx.x & 31;
  const int c = c0 + (lane & 15), half = lane >> 4;
#pragma unroll
  for (int i = 0; i < 8; ++i) p[(r0 + i + 8 * half) * ld + c] = (f16)a[i];
}

// ---------- kernel 0: init global-max cell ----------
__global__ void perf_init_kernel(unsigned* key) {
  if (threadIdx.x == 0) *key = 0x007FFFFFu;  // fkey(-inf)
}

// ---------- kernel 1: global max over norm * k . proj^T ----------
__global__ __launch_bounds__(256) void perf_kmax_kernel(
    const float* __restrict__ k, const float* __restrict__ proj, unsigned* key) {
  extern __shared__ float sp1[];
  float* projs = sp1;              // NF*D_
  float* red   = sp1 + NF * D_;    // 256
  for (int i = threadIdx.x; i < NF * D_; i += 256) projs[i] = proj[i];
  __syncthreads();

  const int pos = blockIdx.x * 256 + threadIdx.x;   // 0 .. B*H*N-1
  const float* kp = k + (size_t)pos * D_;
  float kreg[D_];
#pragma unroll
  for (int d = 0; d < D_; ++d) kreg[d] = kp[d] * NORMZ;

  float mx = -1e30f;
  for (int j = 0; j < NF; ++j) {
    const float* pr = projs + j * D_;
    float dot = 0.f;
#pragma unroll
    for (int d = 0; d < D_; ++d) dot += kreg[d] * pr[d];
    mx = fmaxf(mx, dot);
  }
  red[threadIdx.x] = mx;
  __syncthreads();
  for (int s = 128; s > 0; s >>= 1) {
    if (threadIdx.x < s) red[threadIdx.x] = fmaxf(red[threadIdx.x], red[threadIdx.x + s]);
    __syncthreads();
  }
  if (threadIdx.x == 0) atomicMax(key, fkey(red[0]));
}

// ---------- kernel 2: fused performer causal linear attention ----------
// 1 workgroup per (b,h); 8 wave32s; state S (288x64 f32) register-resident as
// 9 WMMA accumulators per wave; 128-chunk sequential scan; async-LDS
// double-buffered chunk loads.
__global__ __launch_bounds__(256) void perf_attn_kernel(
    const float* __restrict__ q, const float* __restrict__ k,
    const float* __restrict__ v, const float* __restrict__ proj,
    const unsigned* __restrict__ gkey, float* __restrict__ out) {
  extern __shared__ char smem[];
  f16* proj_h = (f16*)smem;             // FP*64
  f16* qh  = proj_h + FP * D_;          // 32*64 (pre-scaled by NORMZ)
  f16* kh  = qh + CH * D_;              // 32*64
  f16* vh  = kh + CH * D_;              // 32*64
  f16* Qc  = vh + CH * D_;              // 32*FP  q' features
  f16* Kc  = Qc + CH * FP;              // 32*FP  k' features
  f16* Sh  = Kc + CH * FP;              // FP*64  f16 snapshot of state S_prev
  f16* Th  = Sh + FP * D_;              // 32*32  causal-masked scores
  float* dd    = (float*)(Th + CH * CH);  // 32*FP  data_dash scratch
  float* Amat  = dd + CH * FP;            // 32*32  partial A, K-half 0
  float* Amat2 = Amat + CH * CH;          // 32*32  partial A, K-half 1
  float* zf    = Amat2 + CH * CH;         // FP     k' feature cumsum state
  float* diagQ = zf + FP;                 // 32
  float* diagK = diagQ + CH;              // 32
  float* mrow  = diagK + CH;              // 32
  float* qrsum = mrow + CH;               // 32
  float* dinv  = qrsum + CH;              // 32
  float* red   = dinv + CH;               // 32*8
  float* qraw  = red + CH * 8;            // 2 * 32*64 raw f32 staging (async)
  float* kraw  = qraw + 2 * CH * D_;      // 2 * 32*64
  float* vraw  = kraw + 2 * CH * D_;      // 2 * 32*64

  const int bh   = blockIdx.x;
  const int t    = threadIdx.x;
  const int wave = t >> 5;
  const float gmax = funkey(*gkey);

  const float* qb = q + (size_t)bh * N_ * D_;
  const float* kb = k + (size_t)bh * N_ * D_;
  const float* vb = v + (size_t)bh * N_ * D_;
  float*       ob = out + (size_t)bh * N_ * D_;

  // proj -> LDS f16, zero-padded to 288 features
  for (int i = t; i < FP * D_; i += 256) {
    int j = i / D_, d = i % D_;
    proj_h[i] = (f16)((j < NF) ? proj[j * D_ + d] : 0.f);
  }
  for (int i = t; i < FP; i += 256) zf[i] = 0.f;

#if USE_ASYNC
  // prefetch chunk 0 into staging buffer 0 (ASYNCcnt-tracked DMA to LDS)
#pragma unroll
  for (int i = 0; i < 2; ++i) {
    int e4 = (t + i * 256) * 4;
    async_cp16(qb + e4, qraw + e4);
    async_cp16(kb + e4, kraw + e4);
    async_cp16(vb + e4, vraw + e4);
  }
#endif

  v8f Sacc[9];
#pragma unroll
  for (int i = 0; i < 9; ++i)
#pragma unroll
    for (int e = 0; e < 8; ++e) Sacc[i][e] = 0.f;
  __syncthreads();

  for (int ch = 0; ch < NCHUNK; ++ch) {
    const int base = ch * CH * D_;

    // (a) snapshot S_prev -> Sh (f16); reset diag accumulators
#pragma unroll
    for (int i = 0; i < 9; ++i) {
      int tile = wave + i * 8, ft = tile >> 2, et = tile & 3;
      st_acc_f16(Sh, D_, ft * 16, et * 16, Sacc[i]);
    }
    if (t < CH) { diagQ[t] = 0.f; diagK[t] = 0.f; }
#if USE_ASYNC
    __builtin_amdgcn_s_wait_asynccnt(0);   // chunk ch staging complete
#endif
    __syncthreads();

#if USE_ASYNC
    // prefetch chunk ch+1 into the other staging buffer, overlapped w/ compute
    if (ch + 1 < NCHUNK) {
      const int nbase = (ch + 1) * CH * D_;
      float* qr = qraw + ((ch + 1) & 1) * (CH * D_);
      float* kr = kraw + ((ch + 1) & 1) * (CH * D_);
      float* vr = vraw + ((ch + 1) & 1) * (CH * D_);
#pragma unroll
      for (int i = 0; i < 2; ++i) {
        int e4 = (t + i * 256) * 4;
        async_cp16(qb + nbase + e4, qr + e4);
        async_cp16(kb + nbase + e4, kr + e4);
        async_cp16(vb + nbase + e4, vr + e4);
      }
    }
    // convert staged raw f32 -> f16 tiles; accumulate 0.5*||.||^2
    {
      const float* qr = qraw + (ch & 1) * (CH * D_);
      const float* kr = kraw + (ch & 1) * (CH * D_);
      const float* vr = vraw + (ch & 1) * (CH * D_);
      for (int i = t; i < CH * D_; i += 256) {
        int r = i >> 6;
        float qv = qr[i] * NORMZ;
        float kv = kr[i] * NORMZ;
        qh[i] = (f16)qv; kh[i] = (f16)kv; vh[i] = (f16)vr[i];
        atomicAdd(&diagQ[r], 0.5f * qv * qv);
        atomicAdd(&diagK[r], 0.5f * kv * kv);
      }
    }
#else
    for (int i = t; i < CH * D_; i += 256) {
      int r = i >> 6;
      float qv = qb[base + i] * NORMZ;
      float kv = kb[base + i] * NORMZ;
      qh[i] = (f16)qv; kh[i] = (f16)kv; vh[i] = (f16)vb[base + i];
      atomicAdd(&diagQ[r], 0.5f * qv * qv);
      atomicAdd(&diagK[r], 0.5f * kv * kv);
    }
#endif
    __syncthreads();

    // (b) dd = qh @ proj^T  (2x18 tiles, K=64)
    for (int tile = wave; tile < 36; tile += 8) {
      int pi = tile & 1, fj = tile >> 1;
      v8f acc = {};
#pragma unroll
      for (int kk = 0; kk < D_; kk += 32) {
        v16h a = ld_frag<true,  false>(qh,     D_, pi * 16, kk);
        v16h b = ld_frag<false, false>(proj_h, D_, fj * 16, kk);
        acc = wmma16(a, b, acc);
      }
      st_acc_f32(dd, FP, pi * 16, fj * 16, acc);
    }
    __syncthreads();

    // q stats: row max, then q' = ratio*(exp(dd - diag - max)+eps); row sums
    {
      int r = t >> 3, s = t & 7;
      float mx = -1e30f;
      for (int j = s * 36; j < s * 36 + 36; ++j)
        if (j < NF) mx = fmaxf(mx, dd[r * FP + j]);
      red[r * 8 + s] = mx;
      __syncthreads();
      if (s == 0) {
        float m2 = red[r * 8];
        for (int i2 = 1; i2 < 8; ++i2) m2 = fmaxf(m2, red[r * 8 + i2]);
        mrow[r] = m2;
      }
      __syncthreads();
      float sub = diagQ[r] + mrow[r], ps = 0.f;
      for (int j = s * 36; j < s * 36 + 36; ++j) {
        float val = 0.f;
        if (j < NF) { val = RATIO * (__expf(dd[r * FP + j] - sub) + KEPS); ps += val; }
        Qc[r * FP + j] = (f16)val;
      }
      red[r * 8 + s] = ps;
      __syncthreads();
      if (s == 0) {
        float sm = 0.f;
        for (int i2 = 0; i2 < 8; ++i2) sm += red[r * 8 + i2];
        qrsum[r] = sm;
      }
    }
    __syncthreads();

    // (c) dd = kh @ proj^T ; k' uses the global max
    for (int tile = wave; tile < 36; tile += 8) {
      int pi = tile & 1, fj = tile >> 1;
      v8f acc = {};
#pragma unroll
      for (int kk = 0; kk < D_; kk += 32) {
        v16h a = ld_frag<true,  false>(kh,     D_, pi * 16, kk);
        v16h b = ld_frag<false, false>(proj_h, D_, fj * 16, kk);
        acc = wmma16(a, b, acc);
      }
      st_acc_f32(dd, FP, pi * 16, fj * 16, acc);
    }
    __syncthreads();
    {
      int r = t >> 3, s = t & 7;
      float sub = diagK[r] + gmax;
      for (int j = s * 36; j < s * 36 + 36; ++j) {
        float val = (j < NF) ? RATIO * (__expf(dd[r * FP + j] - sub) + KEPS) : 0.f;
        Kc[r * FP + j] = (f16)val;
      }
    }
    __syncthreads();

    // (d) all waves: inter-chunk out tile Oacc = Qc @ Sh (K=288, 9 WMMAs)
    const int pi = wave >> 2, ej = wave & 3;
    v8f Oacc = {};
    for (int f0 = 0; f0 < FP; f0 += 32) {
      v16h a = ld_frag<true, false>(Qc, FP, pi * 16, f0);
      v16h b = ld_frag<false, true>(Sh, D_, ej * 16, f0);
      Oacc = wmma16(a, b, Oacc);
    }
    // all waves: one K-half of one A = Qc @ Kc^T tile (load-balanced 8 jobs)
    {
      int tile = wave >> 1, kh2 = wave & 1;
      int api = tile >> 1, apj = tile & 1;
      int fbeg = kh2 ? 160 : 0, fend = kh2 ? FP : 160;
      v8f acc = {};
      for (int f0 = fbeg; f0 < fend; f0 += 32) {
        v16h a = ld_frag<true,  false>(Qc, FP, api * 16, f0);
        v16h b = ld_frag<false, false>(Kc, FP, apj * 16, f0);
        acc = wmma16(a, b, acc);
      }
      st_acc_f32(kh2 ? Amat2 : Amat, CH, api * 16, apj * 16, acc);
    }
    __syncthreads();

    // causal mask -> Th (f16); denom = q'.z_prev + rowsum(T) + AEPS*rowsum(q')
    {
      int r = t >> 3, s = t & 7;
      float asum = 0.f;
      for (int m = s * 4; m < s * 4 + 4; ++m) {
        float x = (m <= r) ? (Amat[r * CH + m] + Amat2[r * CH + m]) : 0.f;
        Th[r * CH + m] = (f16)x;
        asum += x;
      }
      float qz = 0.f;
      for (int j = s * 36; j < s * 36 + 36; ++j)
        qz += (float)Qc[r * FP + j] * zf[j];
      red[r * 8 + s] = asum + qz;
      __syncthreads();
      if (s == 0) {
        float d2 = 0.f;
        for (int i2 = 0; i2 < 8; ++i2) d2 += red[r * 8 + i2];
        dinv[r] = 1.f / (d2 + AEPS * qrsum[r]);
      }
    }
    __syncthreads();

    // (e) Oacc += Th @ Vc (K=32), scale by Dinv, store to global
    {
      v16h a = ld_frag<true, false>(Th, CH, pi * 16, 0);
      v16h b = ld_frag<false, true>(vh, D_, ej * 16, 0);
      Oacc = wmma16(a, b, Oacc);
      const int lane = t & 31;
      const int c = ej * 16 + (lane & 15), half = lane >> 4;
      float* op = ob + (size_t)ch * CH * D_;
#pragma unroll
      for (int i = 0; i < 8; ++i) {
        int r = pi * 16 + i + 8 * half;
        op[r * D_ + c] = Oacc[i] * dinv[r];
      }
    }

    // (f) state update: S += Kc^T @ Vc (one WMMA per owned tile, K=32)
#pragma unroll
    for (int i = 0; i < 9; ++i) {
      int tile = wave + i * 8, ft = tile >> 2, et = tile & 3;
      v16h a = ld_frag<true,  true>(Kc, FP, ft * 16, 0);
      v16h b = ld_frag<false, true>(vh, D_, et * 16, 0);
      Sacc[i] = wmma16(a, b, Sacc[i]);
    }
    // z update (z_prev already consumed in the denominator pass)
    for (int j = t; j < FP; j += 256) {
      float s2 = 0.f;
      for (int m = 0; m < CH; ++m) s2 += (float)Kc[m * FP + j];
      zf[j] += s2;
    }
    __syncthreads();
  }
}

extern "C" void kernel_launch(void* const* d_in, const int* in_sizes, int n_in,
                              void* d_out, int out_size, void* d_ws, size_t ws_size,
                              hipStream_t stream) {
  (void)in_sizes; (void)n_in; (void)out_size; (void)ws_size;
  const float* q    = (const float*)d_in[0];
  const float* k    = (const float*)d_in[1];
  const float* v    = (const float*)d_in[2];
  const float* proj = (const float*)d_in[3];
  float* out = (float*)d_out;
  unsigned* key = (unsigned*)d_ws;

  perf_init_kernel<<<1, 64, 0, stream>>>(key);

  size_t lds1 = (size_t)(NF * D_ + 256) * sizeof(float);
  perf_kmax_kernel<<<(B_ * H_ * N_) / 256, 256, lds1, stream>>>(k, proj, key);

  const size_t f16_elems = (size_t)FP * D_ + 3 * CH * D_ + 2 * CH * FP + FP * D_ + CH * CH;
  const size_t f32_elems = (size_t)CH * FP + 2 * CH * CH + FP + 5 * CH + CH * 8 +
                           6 * CH * D_;   // incl. 2x triple staging buffers
  size_t lds2 = f16_elems * 2 + f32_elems * 4;   // ~217 KB of 320 KB WGP LDS
  perf_attn_kernel<<<B_ * H_, 256, lds2, stream>>>(q, k, v, proj, key, out);
}
